// Hopfield_48679159333472
// MI455X (gfx1250) — matrix operand order repaired and checked by reading, not compiled, not run on hardware
//
#include <hip/hip_runtime.h>
#include <hip/hip_bf16.h>

// ---------------------------------------------------------------------------
// Hopfield attention (B=2, S=2048, D=1024, H=16, HD=64, STEPS=1) for gfx1250.
// f16 WMMA everywhere; TDM (tensor_load_to_lds) stages GEMM weight tiles;
// flash-style online softmax; all transposes hoisted out of hot loops.
// Round 3: GEMM wave tile back to 16x64 (4 accumulators) to eliminate the
// scratch spills seen in round 2; B-fragments consumed immediately.
// ---------------------------------------------------------------------------

typedef __attribute__((ext_vector_type(16))) _Float16 v16h;
typedef __attribute__((ext_vector_type(8)))  float    v8f;
typedef __attribute__((ext_vector_type(4)))  unsigned u32x4;
typedef __attribute__((ext_vector_type(8)))  unsigned u32x8;

#define B_SZ   2
#define S_LEN  2048
#define DIM    1024
#define NH     16
#define HD     64
#define SCALE  0.125f   // 1/sqrt(64)
#define LN_EPS 1e-5f

union U16x8 { uint4 u; _Float16 h[8]; };
union AFrag { v16h v; uint4 u4[2]; };

static __device__ __forceinline__ v8f wmma16(const AFrag& a, const AFrag& b, v8f c) {
  return __builtin_amdgcn_wmma_f32_16x16x32_f16(false, a.v, false, b.v, (short)0, c, false, false);
}

// ---------------------------------------------------------------------------
// TDM: load a 2D tile (32 k-elems x 64 rows, f16) of a 1024x1024 f16 tensor
// into LDS, row-major, 16B pad after each 64B row (LDS row stride 80B).
// D# layout per CDNA5 ISA (group0 128b, group1 256b).
// ---------------------------------------------------------------------------
static __device__ __forceinline__ void tdm_load_tile(const _Float16* gsrc,
                                                     unsigned lds_off) {
  unsigned long long ga = (unsigned long long)(size_t)gsrc;
  u32x4 g0;
  g0[0] = 1u;                                        // count=1, user mode
  g0[1] = lds_off;                                   // lds_addr (bytes)
  g0[2] = (unsigned)(ga & 0xFFFFFFFFull);            // global_addr[31:0]
  g0[3] = (unsigned)((ga >> 32) & 0x01FFFFFFull)     // global_addr[56:32]
          | (2u << 30);                              // type=2 ("image")
  u32x8 g1;
  g1[0] = (1u << 16)                                 // data_size = 2 bytes
        | (1u << 20)                                 // pad_enable
        | (3u << 22)                                 // pad_interval: every 64B
        | (3u << 25);                                // pad_amount: 4 dwords (16B)
  g1[1] = (DIM & 0xFFFFu) << 16;                     // tensor_dim0 lo16 (1024)
  g1[2] = ((DIM >> 16) & 0xFFFFu)                    // tensor_dim0 hi16
        | ((DIM & 0xFFFFu) << 16);                   // tensor_dim1 lo16 (1024)
  g1[3] = ((DIM >> 16) & 0xFFFFu)                    // tensor_dim1 hi16
        | (32u << 16);                               // tile_dim0 = 32 (k)
  g1[4] = 64u;                                       // tile_dim1 = 64, tile_dim2 = 0
  g1[5] = (unsigned)DIM;                             // tensor_dim0_stride lo32
  g1[6] = 0u;
  g1[7] = 0u;
  asm volatile("tensor_load_to_lds %0, %1" :: "s"(g0), "s"(g1) : "memory");
}

// ---------------------------------------------------------------------------
// f32 -> f16 transpose-convert (weights): dst[n][k] = (f16) src[k][n].
// 64x64 tiles through LDS, coalesced loads and 32B-vector stores.
// ---------------------------------------------------------------------------
__global__ void cvtT_kernel(const float* __restrict__ src,
                            _Float16* __restrict__ dst) {
  __shared__ _Float16 T[64][72];
  const int bx = blockIdx.x & 15;          // n-tile
  const int by = blockIdx.x >> 4;          // k-tile
  const int r0 = by * 64, c0 = bx * 64;
  const int r  = threadIdx.x >> 2;         // 0..63
  const int cq = threadIdx.x & 3;          // 0..3 (16-wide chunk)
#pragma unroll
  for (int i = 0; i < 16; ++i)
    T[cq * 16 + i][r] = (_Float16)src[(size_t)(r0 + r) * DIM + c0 + cq * 16 + i];
  __syncthreads();
  U16x8 w0, w1;
#pragma unroll
  for (int i = 0; i < 8; ++i) { w0.h[i] = T[r][cq * 16 + i]; w1.h[i] = T[r][cq * 16 + 8 + i]; }
  _Float16* d = dst + (size_t)(c0 + r) * DIM + r0 + cq * 16;
  *(uint4*)(d)     = w0.u;
  *(uint4*)(d + 8) = w1.u;
}

// ---------------------------------------------------------------------------
// Fused triple LayerNorm: one row per block, stats computed once.
// ---------------------------------------------------------------------------
__global__ void ln3_kernel(const float* __restrict__ data,
                           const float* __restrict__ gk, const float* __restrict__ bk,
                           const float* __restrict__ gq, const float* __restrict__ bq,
                           const float* __restrict__ gv, const float* __restrict__ bv,
                           _Float16* __restrict__ xk, _Float16* __restrict__ xq,
                           _Float16* __restrict__ xv) {
  __shared__ float sw[8], ssw[8];
  const int row = blockIdx.x;
  const float* x = data + (size_t)row * DIM;
  float s = 0.f, ss = 0.f;
  float v0[4];
#pragma unroll
  for (int i = 0; i < 4; ++i) {
    float v = x[threadIdx.x + i * 256];
    v0[i] = v; s += v; ss += v * v;
  }
#pragma unroll
  for (int m = 16; m >= 1; m >>= 1) {
    s  += __shfl_xor(s,  m, 32);
    ss += __shfl_xor(ss, m, 32);
  }
  int wave = threadIdx.x >> 5;
  if ((threadIdx.x & 31) == 0) { sw[wave] = s; ssw[wave] = ss; }
  __syncthreads();
  s = 0.f; ss = 0.f;
#pragma unroll
  for (int w = 0; w < 8; ++w) { s += sw[w]; ss += ssw[w]; }
  float mean = s * (1.0f / DIM);
  float var  = ss * (1.0f / DIM) - mean * mean;
  float rs   = rsqrtf(var + LN_EPS);
#pragma unroll
  for (int i = 0; i < 4; ++i) {
    int c = threadIdx.x + i * 256;
    float nrm = (v0[i] - mean) * rs;
    size_t o = (size_t)row * DIM + c;
    xk[o] = (_Float16)(nrm * gk[c] + bk[c]);
    xq[o] = (_Float16)(nrm * gq[c] + bq[c]);
    xv[o] = (_Float16)(nrm * gv[c] + bv[c]);
  }
}

// ---------------------------------------------------------------------------
// WMMA GEMM: Y[4096,1024] = A(f16) @ W + bias, weights given TRANSPOSED
// (WT[n][k]).  Block 256 thr (8 waves), tile 128(M) x 64(N); wave owns
// 16 rows x 64 cols (4 WMMA per k-step, 4 accumulators -> no spills).
// B tile staged to LDS by the Tensor Data Mover, double-buffered,
// one barrier per k-step.
//   mode 0: f32 row-major.         mode 1: f16 [B,H,S,HD]        (Q)
//   mode 2: f16 both layouts (K).  mode 3: f16 [B,H,HD,S] only   (V)
// ---------------------------------------------------------------------------
__global__ __launch_bounds__(256)
void gemm_kernel(const _Float16* __restrict__ A,
                 const _Float16* __restrict__ WT,
                 const float* __restrict__ bias,
                 void* __restrict__ out, void* __restrict__ out2,
                 int mode) {
  __shared__ __align__(16) _Float16 Wl[2][64 * 40];    // [row n][k], stride 40 f16
  const int wave = threadIdx.x >> 5;
  const int lane = threadIdx.x & 31;
  const int half = lane >> 4;
  const int l16  = lane & 15;
  const int nb = blockIdx.x * 64;
  const int rb = blockIdx.y * 128 + wave * 16;

  v8f z = {0.f,0.f,0.f,0.f,0.f,0.f,0.f,0.f};
  v8f Oc[4] = {z, z, z, z};

  const unsigned lds0 = (unsigned)(size_t)&Wl[0][0];
  const unsigned lds1 = (unsigned)(size_t)&Wl[1][0];
  const _Float16* wtile = WT + (size_t)nb * DIM;       // tile row base

  if (wave == 0) tdm_load_tile(wtile, lds0);

  for (int it = 0; it < DIM / 32; ++it) {
    if (wave == 0) __builtin_amdgcn_s_wait_tensorcnt(0);
    __syncthreads();                                   // tile `it` visible to all
    if (wave == 0 && it + 1 < DIM / 32)
      tdm_load_tile(wtile + (it + 1) * 32, (it & 1) ? lds0 : lds1);

    const _Float16* Wb = (it & 1) ? &Wl[1][0] : &Wl[0][0];

    AFrag af;
    const _Float16* ar = A + (size_t)(rb + l16) * DIM + it * 32;
    af.u4[0] = *(const uint4*)(ar + half * 8);
    af.u4[1] = *(const uint4*)(ar + 16 + half * 8);

#pragma unroll
    for (int nt = 0; nt < 4; ++nt) {
      AFrag bf;
      const _Float16* wp = Wb + (nt * 16 + l16) * 40 + half * 8;
      bf.u4[0] = *(const uint4*)(wp);
      bf.u4[1] = *(const uint4*)(wp + 16);
      Oc[nt] = wmma16(af, bf, Oc[nt]);
    }
  }

  // ---- epilogue: bias add + store (incremental offsets, minimal SALU) ----
#pragma unroll
  for (int nt = 0; nt < 4; ++nt) {
    const int col = nb + nt * 16 + l16;
    const float bvv = bias[col];
    const int row0 = rb + half * 8;
    if (mode == 0) {
      float* p = (float*)out + (size_t)row0 * DIM + col;
#pragma unroll
      for (int j = 0; j < 8; ++j) p[j * DIM] = Oc[nt][j] + bvv;
    } else {
      const int b = row0 >> 11, s0 = row0 & 2047;
      const int h = col >> 6,  hd = col & 63;
      _Float16* pS = (mode != 3)
          ? (_Float16*)out  + (((size_t)(b * NH + h) * S_LEN + s0) * HD) + hd
          : nullptr;
      _Float16* pT = (mode != 1)
          ? (_Float16*)out2 + (((size_t)(b * NH + h) * HD + hd) * S_LEN) + s0
          : nullptr;
#pragma unroll
      for (int j = 0; j < 8; ++j) {
        _Float16 hv = (_Float16)(Oc[nt][j] + bvv);
        if (mode != 3) pS[j * HD] = hv;
        if (mode != 1) pT[j] = hv;
      }
    }
  }
}

// ---------------------------------------------------------------------------
// Flash attention: O = softmax(SCALE * Q K^T) @ Vmat per (b,h).
// Q in [B,H,S,HD]; K in [B,H,S,HD]; Vmat given TRANSPOSED [B,H,HD,S] so all
// WMMA B-fragments are two 16B loads along k.  No barriers; per-wave LDS only
// for the P-matrix fragment reshape.  Block 256 thr (8 waves x 16 q-rows).
// outMode 0: f16 [B,H,S,HD] (Hopfield update, Vmat = K^T).
// outMode 1: f16 merged [B,S,D] (retrieval, Vmat = V^T).
// ---------------------------------------------------------------------------
__global__ __launch_bounds__(256)
void attn_kernel(const _Float16* __restrict__ Qp,
                 const _Float16* __restrict__ Kp,
                 const _Float16* __restrict__ VTp,
                 _Float16* __restrict__ Op, int outMode) {
  __shared__ __align__(16) _Float16 Pbuf[8][16 * 32];

  const int bh   = blockIdx.y;                // 0..31
  const int wave = threadIdx.x >> 5;
  const int lane = threadIdx.x & 31;
  const int half = lane >> 4;
  const int l16  = lane & 15;
  const int qrow = blockIdx.x * 128 + wave * 16;

  const _Float16* Q  = Qp  + (size_t)bh * S_LEN * HD;
  const _Float16* K  = Kp  + (size_t)bh * S_LEN * HD;
  const _Float16* VT = VTp + (size_t)bh * HD * S_LEN;

  AFrag a0, a1;   // loop-invariant Q fragments (hd 0..31, 32..63)
  {
    const _Float16* qr = Q + (size_t)(qrow + l16) * HD;
    a0.u4[0] = *(const uint4*)(qr +      half * 8);
    a0.u4[1] = *(const uint4*)(qr + 16 + half * 8);
    a1.u4[0] = *(const uint4*)(qr + 32 + half * 8);
    a1.u4[1] = *(const uint4*)(qr + 48 + half * 8);
  }

  v8f z = {0.f,0.f,0.f,0.f,0.f,0.f,0.f,0.f};
  v8f Oc[4] = {z, z, z, z};
  float mrow[8], lrow[8];
#pragma unroll
  for (int j = 0; j < 8; ++j) { mrow[j] = -3.0e38f; lrow[j] = 0.f; }

  for (int kt = 0; kt < S_LEN; kt += 32) {
    // ---- scores: s = Q(16x64) K^T(64x32), K B-frags contiguous from global
    v8f s0 = z, s1 = z;
    {
      const _Float16* kr0 = K + (size_t)(kt + l16) * HD + half * 8;
      const _Float16* kr1 = K + (size_t)(kt + 16 + l16) * HD + half * 8;
      AFrag b00, b01, b10, b11;
      b00.u4[0] = *(const uint4*)(kr0);      b00.u4[1] = *(const uint4*)(kr0 + 16);
      b01.u4[0] = *(const uint4*)(kr0 + 32); b01.u4[1] = *(const uint4*)(kr0 + 48);
      b10.u4[0] = *(const uint4*)(kr1);      b10.u4[1] = *(const uint4*)(kr1 + 16);
      b11.u4[0] = *(const uint4*)(kr1 + 32); b11.u4[1] = *(const uint4*)(kr1 + 48);
      s0 = wmma16(a0, b00, s0);  s0 = wmma16(a1, b01, s0);
      s1 = wmma16(a0, b10, s1);  s1 = wmma16(a1, b11, s1);
    }

    // ---- online softmax over this 32-key tile ----
#pragma unroll
    for (int j = 0; j < 8; ++j) {
      float e0 = s0[j] * SCALE;
      float e1 = s1[j] * SCALE;
      float t = fmaxf(e0, e1);
#pragma unroll
      for (int m = 1; m <= 8; m <<= 1) t = fmaxf(t, __shfl_xor(t, m, 32));
      float mnew  = fmaxf(mrow[j], t);
      float alpha = __expf(mrow[j] - mnew);
      float p0 = __expf(e0 - mnew);
      float p1 = __expf(e1 - mnew);
      float rsum = p0 + p1;
#pragma unroll
      for (int m = 1; m <= 8; m <<= 1) rsum += __shfl_xor(rsum, m, 32);
      lrow[j] = lrow[j] * alpha + rsum;
      mrow[j] = mnew;
#pragma unroll
      for (int nt = 0; nt < 4; ++nt) Oc[nt][j] *= alpha;
      int prow = j + half * 8;
      Pbuf[wave][prow * 32 + l16]      = (_Float16)p0;
      Pbuf[wave][prow * 32 + 16 + l16] = (_Float16)p1;
    }

    // ---- O += P(16x32) V(32x64), V^T B-frags contiguous from global ----
    AFrag pf;
    {
      const _Float16* pb = &Pbuf[wave][0] + l16 * 32;
      pf.u4[0] = *(const uint4*)(pb +      half * 8);
      pf.u4[1] = *(const uint4*)(pb + 16 + half * 8);
    }
#pragma unroll
    for (int nt = 0; nt < 4; ++nt) {
      AFrag vb;
      const _Float16* vp = VT + (size_t)(nt * 16 + l16) * S_LEN + kt + half * 8;
      vb.u4[0] = *(const uint4*)(vp);
      vb.u4[1] = *(const uint4*)(vp + 16);
      Oc[nt] = wmma16(pf, vb, Oc[nt]);
    }
  }

  // ---- epilogue: normalize and store ----
#pragma unroll
  for (int nt = 0; nt < 4; ++nt) {
    const int hd  = nt * 16 + l16;
    const int row0 = qrow + half * 8;
    _Float16* p;
    int stride;
    if (outMode == 0) {
      p = Op + (size_t)bh * S_LEN * HD + (size_t)row0 * HD + hd;   // [B,H,S,HD]
      stride = HD;
    } else {
      int b = bh >> 4, h = bh & 15;
      p = Op + ((size_t)(b * S_LEN + row0)) * DIM + h * HD + hd;   // [B,S,D]
      stride = DIM;
    }
#pragma unroll
    for (int j = 0; j < 8; ++j)
      p[j * stride] = (_Float16)(Oc[nt][j] / lrow[j]);
  }
}

// ---------------------------------------------------------------------------
// Host orchestration
// ---------------------------------------------------------------------------
extern "C" void kernel_launch(void* const* d_in, const int* in_sizes, int n_in,
                              void* d_out, int out_size, void* d_ws, size_t ws_size,
                              hipStream_t stream) {
  const float* data = (const float*)d_in[0];
  const float* g_k  = (const float*)d_in[1];
  const float* b_k  = (const float*)d_in[2];
  const float* g_q  = (const float*)d_in[3];
  const float* b_q  = (const float*)d_in[4];
  const float* g_v  = (const float*)d_in[5];
  const float* b_v  = (const float*)d_in[6];
  const float* Wq   = (const float*)d_in[7];
  const float* bq   = (const float*)d_in[8];
  const float* Wk   = (const float*)d_in[9];
  const float* bk   = (const float*)d_in[10];
  const float* Wv   = (const float*)d_in[11];
  const float* bv   = (const float*)d_in[12];
  const float* Wo   = (const float*)d_in[13];
  const float* bo   = (const float*)d_in[14];

  const size_t SZ_W = (size_t)DIM * DIM * sizeof(_Float16);          // 2 MiB
  const size_t SZ_T = (size_t)B_SZ * S_LEN * DIM * sizeof(_Float16); // 8 MiB

  char* ws = (char*)d_ws;
  size_t off = 0;
  _Float16* WqT = (_Float16*)(ws + off); off += SZ_W;   // transposed f16 weights
  _Float16* WkT = (_Float16*)(ws + off); off += SZ_W;
  _Float16* WvT = (_Float16*)(ws + off); off += SZ_W;
  _Float16* WoT = (_Float16*)(ws + off); off += SZ_W;
  _Float16* xqn = (_Float16*)(ws + off); off += SZ_T;
  _Float16* xkn = (_Float16*)(ws + off); off += SZ_T;
  _Float16* xvn = (_Float16*)(ws + off); off += SZ_T;
  _Float16* Qh  = (_Float16*)(ws + off); off += SZ_T;   // [B,H,S,HD]
  _Float16* Kh  = (_Float16*)(ws + off); off += SZ_T;   // [B,H,S,HD]
  _Float16* KhT = (_Float16*)(ws + off); off += SZ_T;   // [B,H,HD,S]
  _Float16* VhT = (_Float16*)(ws + off); off += SZ_T;   // [B,H,HD,S]
  _Float16* Q2h = (_Float16*)(ws + off); off += SZ_T;   // [B,H,S,HD]
  _Float16* Am  = (_Float16*)(ws + off); off += SZ_T;   // merged [B,S,D]

  cvtT_kernel<<<256, 256, 0, stream>>>(Wq, WqT);
  cvtT_kernel<<<256, 256, 0, stream>>>(Wk, WkT);
  cvtT_kernel<<<256, 256, 0, stream>>>(Wv, WvT);
  cvtT_kernel<<<256, 256, 0, stream>>>(Wo, WoT);

  ln3_kernel<<<B_SZ * S_LEN, 256, 0, stream>>>(data, g_k, b_k, g_q, b_q, g_v, b_v,
                                               xkn, xqn, xvn);

  dim3 ggrid(DIM / 64, (B_SZ * S_LEN) / 128);           // (16, 32)
  gemm_kernel<<<ggrid, 256, 0, stream>>>(xqn, WqT, bq, (void*)Qh, nullptr, 1);
  gemm_kernel<<<ggrid, 256, 0, stream>>>(xkn, WkT, bk, (void*)Kh, (void*)KhT, 2);
  gemm_kernel<<<ggrid, 256, 0, stream>>>(xvn, WvT, bv, nullptr, (void*)VhT, 3);

  dim3 agrid(S_LEN / 128, B_SZ * NH);                   // (16, 32)
  // Hopfield update step: q <- softmax(beta q K^T) K
  attn_kernel<<<agrid, 256, 0, stream>>>(Qh, Kh, KhT, Q2h, 0);
  // Retrieval: out = softmax(beta q K^T) V, merged-head layout
  attn_kernel<<<agrid, 256, 0, stream>>>(Q2h, Kh, VhT, Am, 1);

  gemm_kernel<<<ggrid, 256, 0, stream>>>(Am, WoT, bo, d_out, nullptr, 0);
}
